// DepthNetv3_80642305950154
// MI455X (gfx1250) — matrix operand order repaired and checked by reading, not compiled, not run on hardware
//
#include <hip/hip_runtime.h>
#include <hip/hip_bf16.h>
#include <math.h>

// Problem constants (from reference setup_inputs)
static constexpr int V  = 5;     // total views
static constexpr int NS = 4;     // source views
static constexpr int C  = 32;    // channels
static constexpr int H  = 128;
static constexpr int W  = 160;
static constexpr int D  = 48;    // depth planes
static constexpr int HW = H * W; // 20480

typedef __attribute__((ext_vector_type(16))) _Float16 v16h;
typedef __attribute__((ext_vector_type(8)))  float    v8f;

// ---------------------------------------------------------------------------
// Kernel 1: features (V,1,C,H,W) -> channel-last (V, H*W, C) for float4 gathers
// ---------------------------------------------------------------------------
__global__ void k_transpose(const float* __restrict__ feat, float* __restrict__ out) {
  int idx = blockIdx.x * blockDim.x + threadIdx.x;
  if (idx >= V * C * HW) return;
  int w_ = idx % W;
  int t  = idx / W;
  int h  = t % H;  t /= H;
  int c  = t % C;
  int v  = t / C;
  out[((size_t)v * HW + h * W + w_) * C + c] = feat[idx];
}

// ---------------------------------------------------------------------------
// Kernel 2: camera math. proj = fuse(src) @ inv(fuse(ref)); store rot(9)+trans(3)
// ---------------------------------------------------------------------------
__device__ inline void fuse4(const float* pm, float* F) {
  const float* E = pm;       // p[:,0]
  const float* K = pm + 16;  // p[:,1]
#pragma unroll
  for (int r = 0; r < 3; ++r)
#pragma unroll
    for (int c = 0; c < 4; ++c)
      F[r * 4 + c] = K[r * 4 + 0] * E[0 + c] + K[r * 4 + 1] * E[4 + c] + K[r * 4 + 2] * E[8 + c];
  F[12] = E[12]; F[13] = E[13]; F[14] = E[14]; F[15] = E[15];
}

__global__ void k_proj(const float* __restrict__ pm, float* __restrict__ projRT) {
  if (threadIdx.x != 0 || blockIdx.x != 0) return;
  float Fr[16];
  fuse4(pm, Fr);  // fused ref (view 0), last row (0,0,0,1)
  float M[9] = {Fr[0], Fr[1], Fr[2], Fr[4], Fr[5], Fr[6], Fr[8], Fr[9], Fr[10]};
  float t0 = Fr[3], t1 = Fr[7], t2 = Fr[11];
  float c00 =  M[4] * M[8] - M[5] * M[7];
  float c01 = -(M[3] * M[8] - M[5] * M[6]);
  float c02 =  M[3] * M[7] - M[4] * M[6];
  float det = M[0] * c00 + M[1] * c01 + M[2] * c02;
  float id  = 1.0f / det;
  float Mi[9];
  Mi[0] = c00 * id;
  Mi[1] = -(M[1] * M[8] - M[2] * M[7]) * id;
  Mi[2] =  (M[1] * M[5] - M[2] * M[4]) * id;
  Mi[3] = c01 * id;
  Mi[4] =  (M[0] * M[8] - M[2] * M[6]) * id;
  Mi[5] = -(M[0] * M[5] - M[2] * M[3]) * id;
  Mi[6] = c02 * id;
  Mi[7] = -(M[0] * M[7] - M[1] * M[6]) * id;
  Mi[8] =  (M[0] * M[4] - M[1] * M[3]) * id;
  float R[16];  // inv(fused ref): [[Mi, -Mi t],[0,0,0,1]]
  R[0] = Mi[0]; R[1] = Mi[1]; R[2]  = Mi[2]; R[3]  = -(Mi[0] * t0 + Mi[1] * t1 + Mi[2] * t2);
  R[4] = Mi[3]; R[5] = Mi[4]; R[6]  = Mi[5]; R[7]  = -(Mi[3] * t0 + Mi[4] * t1 + Mi[5] * t2);
  R[8] = Mi[6]; R[9] = Mi[7]; R[10] = Mi[8]; R[11] = -(Mi[6] * t0 + Mi[7] * t1 + Mi[8] * t2);
  R[12] = 0.f; R[13] = 0.f; R[14] = 0.f; R[15] = 1.f;
  for (int v = 1; v < V; ++v) {
    float Fs[16];
    fuse4(pm + v * 32, Fs);
    float P[16];
#pragma unroll
    for (int r = 0; r < 4; ++r)
#pragma unroll
      for (int c = 0; c < 4; ++c) {
        float s = 0.f;
#pragma unroll
        for (int k = 0; k < 4; ++k) s += Fs[r * 4 + k] * R[k * 4 + c];
        P[r * 4 + c] = s;
      }
    float* o = projRT + (v - 1) * 12;
    o[0] = P[0];  o[1] = P[1];  o[2] = P[2];
    o[3] = P[4];  o[4] = P[5];  o[5] = P[6];
    o[6] = P[8];  o[7] = P[9];  o[8] = P[10];
    o[9] = P[3];  o[10] = P[7]; o[11] = P[11];
  }
}

// ---------------------------------------------------------------------------
// Kernel 3: homography warp + bilinear sample + channel dot -> sim volumes
// one thread per (src_view, pixel); loops 48 depths; float4 gathers (128B rows)
// ---------------------------------------------------------------------------
__device__ inline float dot32(const float4* __restrict__ a, const float4* __restrict__ r) {
  float s = 0.f;
#pragma unroll
  for (int q = 0; q < 8; ++q) {
    float4 x = a[q], y = r[q];
    s += x.x * y.x + x.y * y.y + x.z * y.z + x.w * y.w;
  }
  return s;
}

__device__ inline float corner_dot(const float* __restrict__ src, const float4* __restrict__ rr,
                                   float xi, float yi) {
  bool valid = (xi >= 0.f) && (xi <= (float)(W - 1)) && (yi >= 0.f) && (yi <= (float)(H - 1));
  int xc = (int)fminf(fmaxf(xi, 0.f), (float)(W - 1));
  int yc = (int)fminf(fmaxf(yi, 0.f), (float)(H - 1));
  const float4* a = reinterpret_cast<const float4*>(src + ((size_t)yc * W + xc) * C);
  float d = dot32(a, rr);
  return valid ? d : 0.f;
}

__global__ void k_warp_sim(const float* __restrict__ featCL, const float* __restrict__ projRT,
                           const float* __restrict__ depth_vals, float* __restrict__ simvol) {
  int tid = blockIdx.x * blockDim.x + threadIdx.x;
  if (tid >= NS * HW) return;
  int i  = tid / HW;
  int p  = tid - i * HW;
  int h  = p / W;
  int w_ = p - h * W;

  // reference feature vector for this pixel (32 floats, kept in registers)
  const float4* refp = reinterpret_cast<const float4*>(featCL + (size_t)p * C);
  float4 rr[8];
#pragma unroll
  for (int q = 0; q < 8; ++q) rr[q] = refp[q];

  const float* RT = projRT + i * 12;
  float fx = (float)w_, fy = (float)h;
  float rx = RT[0] * fx + RT[1] * fy + RT[2];
  float ry = RT[3] * fx + RT[4] * fy + RT[5];
  float rz = RT[6] * fx + RT[7] * fy + RT[8];
  float tx = RT[9], ty = RT[10], tz = RT[11];

  const float* src = featCL + (size_t)(i + 1) * HW * C;
  float* out = simvol + (size_t)i * D * HW + p;
  const float inv_c = 1.0f / (float)C;

  for (int d = 0; d < D; ++d) {
    float dep = depth_vals[d * HW + p];
    float X = rx * dep + tx;
    float Y = ry * dep + ty;
    float Z = rz * dep + tz;
    float px = X / Z, py = Y / Z;
    float x0 = floorf(px), y0 = floorf(py);
    float wx1 = px - x0, wx0 = 1.f - wx1;
    float wy1 = py - y0, wy0 = 1.f - wy1;
    float s = corner_dot(src, rr, x0,       y0)       * (wx0 * wy0)
            + corner_dot(src, rr, x0 + 1.f, y0)       * (wx1 * wy0)
            + corner_dot(src, rr, x0,       y0 + 1.f) * (wx0 * wy1)
            + corner_dot(src, rr, x0 + 1.f, y0 + 1.f) * (wx1 * wy1);
    out[(size_t)d * HW] = s * inv_c;
  }
}

// ---------------------------------------------------------------------------
// Kernel 4: pixelwise net via WMMA. y = w1(8x16) @ relu(w0*s+b0)(16 x 16pixels)
// One wave handles 16 pixels; loop over 48 depths; one v_wmma per depth.
// A: w1 padded to 16x32 f16. B: per-pixel activations, K rows 16..31 = 0.
// C-fragment: lane n (<16) holds outputs m=0..7 of pixel n in cfrag[0..7].
// sigmoid is strictly increasing => max_d sigmoid(z_d) == sigmoid(max_d z_d):
// track zmax in the loop, single exp+div at the end (removes 47 exp chains).
// ---------------------------------------------------------------------------
__global__ void k_vw_wmma(const float* __restrict__ simvol,
                          const float* __restrict__ w0, const float* __restrict__ b0,
                          const float* __restrict__ w1, const float* __restrict__ b1,
                          const float* __restrict__ w2, const float* __restrict__ b2,
                          float* __restrict__ vw, float* __restrict__ out_vw) {
  int gtid = blockIdx.x * blockDim.x + threadIdx.x;
  int wave = gtid >> 5;
  int lane = gtid & 31;
  const int nwaves = NS * (HW / 16);
  if (wave >= nwaves) return;          // wave-uniform guard
  int i  = wave / (HW / 16);
  int pg = wave - i * (HW / 16);
  int p0 = pg * 16;
  int ln = lane & 15;
  bool lo = lane < 16;

  // A fragment: contiguous k order, same as B. lanes<16: row M=ln, K=0..15;
  // lanes>=16: row M=ln, K=16..31. Rows >=8 and K>=16 are zero-pad.
  v16h a;
#pragma unroll
  for (int k = 0; k < 8; ++k) {
    bool av = (ln < 8) && lo;
    a[k]     = (_Float16)(av ? w1[ln * 16 + k] : 0.0f);
    a[k + 8] = (_Float16)(av ? w1[ln * 16 + k + 8] : 0.0f);
  }

  float w0r[16], b0r[16];
#pragma unroll
  for (int k = 0; k < 16; ++k) { w0r[k] = w0[k]; b0r[k] = b0[k]; }
  float b1r[8], w2r[8];
#pragma unroll
  for (int m = 0; m < 8; ++m) { b1r[m] = b1[m]; w2r[m] = w2[m]; }
  float b2v = b2[0];

  int pix = p0 + ln;
  const float* simbase = simvol + (size_t)i * D * HW;
  float zmax = -INFINITY;

  for (int d = 0; d < D; ++d) {
    float s = simbase[(size_t)d * HW + pix];  // lanes 0-15 coalesced; hi lanes dup
    // B fragment: 32x16 f16; lanes<16 carry K=0..15 of column=lane, hi lanes = K pad (0)
    v16h b;
#pragma unroll
    for (int k = 0; k < 16; ++k) {
      float x = fmaxf(w0r[k] * s + b0r[k], 0.0f);
      b[k] = lo ? (_Float16)x : (_Float16)0.0f;
    }
    v8f cfrag = {};
    cfrag = __builtin_amdgcn_wmma_f32_16x16x32_f16(false, a, false, b,
                                                   (short)0, cfrag, false, false);
    float z = b2v;
#pragma unroll
    for (int m = 0; m < 8; ++m) z += w2r[m] * fmaxf(cfrag[m] + b1r[m], 0.0f);
    zmax = fmaxf(zmax, z);
  }
  float pmax = 1.0f / (1.0f + expf(-zmax));   // sigmoid of max == max of sigmoid
  if (lo) {
    vw[(size_t)i * HW + pix]     = pmax;
    out_vw[(size_t)i * HW + pix] = pmax;
  }
}

// ---------------------------------------------------------------------------
// Kernel 5: weighted fusion of sim volumes
// ---------------------------------------------------------------------------
__global__ void k_fuse(const float* __restrict__ simvol, const float* __restrict__ vw,
                       float* __restrict__ fused) {
  int idx = blockIdx.x * blockDim.x + threadIdx.x;
  if (idx >= D * HW) return;
  int p = idx % HW;
  float num = 0.f, den = 1e-5f;
#pragma unroll
  for (int i = 0; i < NS; ++i) {
    float v = vw[i * HW + p];
    num += simvol[(size_t)i * D * HW + idx] * v;
    den += v;
  }
  fused[idx] = num / den;
}

// ---------------------------------------------------------------------------
// Kernel 6: 3x3x3 SAME conv (1 ch) + softmax over depth + argmax/gather/conf
// one thread per pixel; sliding 3-plane window of 9 spatial taps each
// ---------------------------------------------------------------------------
__device__ inline void load_plane(const float* __restrict__ fused, int dd, int h, int w_,
                                  float* pl) {
  if (dd < 0 || dd >= D) {
#pragma unroll
    for (int j = 0; j < 9; ++j) pl[j] = 0.f;
    return;
  }
  const float* base = fused + (size_t)dd * HW;
#pragma unroll
  for (int kh = 0; kh < 3; ++kh)
#pragma unroll
    for (int kw = 0; kw < 3; ++kw) {
      int hh = h + kh - 1, ww = w_ + kw - 1;
      bool ok = (hh >= 0) && (hh < H) && (ww >= 0) && (ww < W);
      pl[kh * 3 + kw] = ok ? base[hh * W + ww] : 0.f;
    }
}

__global__ void k_reg_softmax(const float* __restrict__ fused, const float* __restrict__ reg_w,
                              const float* __restrict__ reg_b, const float* __restrict__ depth_vals,
                              float* __restrict__ out_depth, float* __restrict__ out_conf,
                              float* __restrict__ out_prob) {
  int p = blockIdx.x * blockDim.x + threadIdx.x;
  if (p >= HW) return;
  int h = p / W, w_ = p - h * W;

  float kw[27];
#pragma unroll
  for (int j = 0; j < 27; ++j) kw[j] = reg_w[j];
  float kb = reg_b[0];

  float pA[9], pB[9], pC[9];
  load_plane(fused, -1, h, w_, pA);
  load_plane(fused, 0, h, w_, pB);

  float cost[D];
  for (int d = 0; d < D; ++d) {
    load_plane(fused, d + 1, h, w_, pC);
    float s = kb;
#pragma unroll
    for (int j = 0; j < 9; ++j)
      s += kw[j] * pA[j] + kw[9 + j] * pB[j] + kw[18 + j] * pC[j];
    cost[d] = s;
#pragma unroll
    for (int j = 0; j < 9; ++j) { pA[j] = pB[j]; pB[j] = pC[j]; }
  }

  float m = -INFINITY;
#pragma unroll
  for (int d = 0; d < D; ++d) m = fmaxf(m, cost[d]);
  float sum = 0.f;
#pragma unroll
  for (int d = 0; d < D; ++d) sum += expf(cost[d] - m);
  float inv = 1.0f / sum;

  float best = -1.f;
  int bidx = 0;
  for (int d = 0; d < D; ++d) {
    float pr = expf(cost[d] - m) * inv;
    out_prob[(size_t)d * HW + p] = pr;
    if (pr > best) { best = pr; bidx = d; }  // strict > : first max index (jnp.argmax)
  }
  out_depth[p] = depth_vals[(size_t)bidx * HW + p];
  out_conf[p]  = best;
}

// ---------------------------------------------------------------------------
extern "C" void kernel_launch(void* const* d_in, const int* in_sizes, int n_in,
                              void* d_out, int out_size, void* d_ws, size_t ws_size,
                              hipStream_t stream) {
  const float* features   = (const float*)d_in[0];   // (5,1,32,128,160)
  const float* proj_mats  = (const float*)d_in[1];   // (1,5,2,4,4)
  const float* depth_vals = (const float*)d_in[2];   // (1,48,128,160)
  const float* w0 = (const float*)d_in[3];
  const float* b0 = (const float*)d_in[4];
  const float* w1 = (const float*)d_in[5];
  const float* b1 = (const float*)d_in[6];
  const float* w2 = (const float*)d_in[7];
  const float* b2 = (const float*)d_in[8];
  const float* reg_w = (const float*)d_in[9];
  const float* reg_b = (const float*)d_in[10];

  // workspace layout (floats)
  float* wsf    = (float*)d_ws;
  float* projRT = wsf;                                  // 48 (padded to 64)
  float* featCL = wsf + 64;                             // 5*HW*C
  float* simvol = featCL + (size_t)V * HW * C;          // NS*D*HW
  float* vw     = simvol + (size_t)NS * D * HW;         // NS*HW
  float* fused  = vw + (size_t)NS * HW;                 // D*HW

  // output layout (return order: depth, conf, prob, view_weights)
  float* out       = (float*)d_out;
  float* out_depth = out;
  float* out_conf  = out + HW;
  float* out_prob  = out + 2 * HW;
  float* out_vw    = out + 2 * HW + D * HW;

  {
    int n = V * C * HW;
    k_transpose<<<(n + 255) / 256, 256, 0, stream>>>(features, featCL);
  }
  k_proj<<<1, 32, 0, stream>>>(proj_mats, projRT);
  {
    int n = NS * HW;
    k_warp_sim<<<(n + 255) / 256, 256, 0, stream>>>(featCL, projRT, depth_vals, simvol);
  }
  {
    int nwaves = NS * (HW / 16);               // 5120 waves
    int nthreads = nwaves * 32;
    k_vw_wmma<<<(nthreads + 255) / 256, 256, 0, stream>>>(simvol, w0, b0, w1, b1, w2, b2,
                                                          vw, out_vw);
  }
  {
    int n = D * HW;
    k_fuse<<<(n + 255) / 256, 256, 0, stream>>>(simvol, vw, fused);
  }
  k_reg_softmax<<<(HW + 255) / 256, 256, 0, stream>>>(fused, reg_w, reg_b, depth_vals,
                                                      out_depth, out_conf, out_prob);
}
